// DeltaBlock_78245714199355
// MI455X (gfx1250) — matrix operand order repaired and checked by reading, not compiled, not run on hardware
//
#include <hip/hip_runtime.h>
#include <stdint.h>
#include <math.h>

typedef __attribute__((ext_vector_type(16))) _Float16 v16h;
typedef __attribute__((ext_vector_type(8)))  float    v8f;

#define D_MODEL 512
#define HEADS   8
#define DH      64
#define BATCH   2
#define SEQ     1024
#define NROWS   (BATCH*SEQ)
#define HIDDEN  2048

__device__ __forceinline__ float sigmoid_f(float x){ return 1.0f/(1.0f+__expf(-x)); }
__device__ __forceinline__ float softplus_f(float x){ return log1pf(__expf(x)); }

// Low 32 bits of a generic pointer to __shared__ = wave-relative LDS byte address
__device__ __forceinline__ unsigned lds_addr_of(const void* p) {
  return (unsigned)(uintptr_t)p;
}
// Async DMA 16 bytes global -> LDS (ASYNCcnt-tracked), saddr addressing mode
__device__ __forceinline__ void async_copy_b128(unsigned lds_byte_addr,
                                                unsigned global_byte_off,
                                                const void* sbase) {
  asm volatile("global_load_async_to_lds_b128 %0, %1, %2"
               :: "v"(lds_byte_addr), "v"(global_byte_off), "s"(sbase)
               : "memory");
}
__device__ __forceinline__ void wait_async0() {
  asm volatile("s_wait_asynccnt 0" ::: "memory");
}
__device__ __forceinline__ void wait_ds0() {
  asm volatile("s_wait_dscnt 0" ::: "memory");
}

// ---------------- LayerNorm: one 512-wide row per 256-thread block ----------
__global__ void ln512_kernel(const float* __restrict__ x, const float* __restrict__ w,
                             const float* __restrict__ b, float* __restrict__ out) {
  __shared__ float red[256];
  const int row = blockIdx.x, tid = threadIdx.x;
  const float* xr = x + (size_t)row * D_MODEL;
  float a0 = xr[tid], a1 = xr[tid + 256];
  red[tid] = a0 + a1;
  __syncthreads();
  for (int off = 128; off > 0; off >>= 1) { if (tid < off) red[tid] += red[tid + off]; __syncthreads(); }
  const float mu = red[0] * (1.0f / D_MODEL);
  __syncthreads();
  float d0 = a0 - mu, d1 = a1 - mu;
  red[tid] = d0 * d0 + d1 * d1;
  __syncthreads();
  for (int off = 128; off > 0; off >>= 1) { if (tid < off) red[tid] += red[tid + off]; __syncthreads(); }
  const float inv = rsqrtf(red[0] * (1.0f / D_MODEL) + 1e-5f);
  float* orow = out + (size_t)row * D_MODEL;
  orow[tid]       = d0 * inv * w[tid]       + b[tid];
  orow[tid + 256] = d1 * inv * w[tid + 256] + b[tid + 256];
}

// ---------------- elementwise tanh ------------------------------------------
__global__ void tanh_kernel(const float* __restrict__ in, float* __restrict__ out, int n) {
  for (int i = blockIdx.x * blockDim.x + threadIdx.x; i < n; i += gridDim.x * blockDim.x)
    out[i] = tanhf(in[i]);
}

// ---------------- proxy linear: out = post((0.5 - meanL1(tx,tw)) * gain) ----
// tx:[N,Din] tw:[M,Din] out:[N,M]; mode 0=none 1=tanh 2=silu
__global__ void proxy_kernel(const float* __restrict__ tx, const float* __restrict__ tw,
                             const float* __restrict__ gain, float* __restrict__ out,
                             int N, int M, int Din, int mode) {
  __shared__ float xs[16][33];
  __shared__ float ws_[16][33];
  const int txi = threadIdx.x, tyi = threadIdx.y;
  const int tid = tyi * 16 + txi;
  const int n0 = blockIdx.y * 16, m0 = blockIdx.x * 16;
  float acc = 0.0f;
  for (int k0 = 0; k0 < Din; k0 += 32) {
    __syncthreads();
    for (int i = tid; i < 16 * 32; i += 256) {
      int r = i >> 5, c = i & 31;
      xs[r][c]  = tx[(size_t)(n0 + r) * Din + k0 + c];
      ws_[r][c] = tw[(size_t)(m0 + r) * Din + k0 + c];
    }
    // hint the next tile into cache while we chew on this one (global_prefetch_b8)
    if (k0 + 32 < Din) {
      __builtin_prefetch(&tx[(size_t)(n0 + tyi) * Din + k0 + 32 + txi * 2], 0, 3);
      __builtin_prefetch(&tw[(size_t)(m0 + tyi) * Din + k0 + 32 + txi * 2], 0, 3);
    }
    __syncthreads();
    #pragma unroll 8
    for (int c = 0; c < 32; ++c) acc += fabsf(xs[tyi][c] - ws_[txi][c]);
  }
  float val = (0.5f - acc / (float)Din) * gain[m0 + txi];
  if (mode == 1) val = tanhf(val);
  else if (mode == 2) val = val * sigmoid_f(val);
  out[(size_t)(n0 + tyi) * M + m0 + txi] = val;
}

// ---------------- attention: streaming student-t kernel + WMMA PV -----------
// q,k,v laid out [B*T, D_MODEL], head h occupies cols h*DH..h*DH+63
__global__ void attn_kernel(const float* __restrict__ q, const float* __restrict__ k,
                            const float* __restrict__ v, const float* __restrict__ gamma,
                            const float* __restrict__ rho, float* __restrict__ out) {
  __shared__ __align__(16) float q_lds[16][DH + 1];
  __shared__ __align__(16) float k_lds[32][DH];
  __shared__ __align__(16) float v_lds[32][DH];
  __shared__ float denom[32];
  const int lane = threadIdx.x;                 // wave32
  const int bh = blockIdx.y, b = bh / HEADS, h = bh % HEADS;
  const int i0 = blockIdx.x * 16;
  const int m  = lane & 15;
  const int hi = lane >> 4;
  const float gp = softplus_f(gamma[h]);
  const float rp = softplus_f(rho[h]);
  const float* qb = q + (size_t)(b * SEQ) * D_MODEL + h * DH;
  const float* kb = k + (size_t)(b * SEQ) * D_MODEL + h * DH;
  const float* vb = v + (size_t)(b * SEQ) * D_MODEL + h * DH;
  for (int i = lane; i < 16 * DH; i += 32) {
    int r = i / DH, d = i % DH;
    q_lds[r][d] = qb[(size_t)(i0 + r) * D_MODEL + d];
  }
  v8f c0 = {}, c1 = {}, c2 = {}, c3 = {};
  float den = 0.0f;
  const int qi = i0 + m;
  const int jmax = i0 + 15;
  const unsigned k_lds0 = lds_addr_of(&k_lds[0][0]);
  const unsigned v_lds0 = lds_addr_of(&v_lds[0][0]);
  for (int j0 = 0; j0 <= jmax; j0 += 32) {
    __syncthreads();
    wait_ds0();   // previous tile's LDS reads drained before overwrite
    // async DMA of the 32x64 f32 K and V tiles straight into LDS (16B/lane chunks)
    // chunk c (of 512): row r = c>>4, col4 = (c&15)*4
    #pragma unroll
    for (int it = 0; it < 16; ++it) {
      const int c = it * 32 + lane;
      const int r = c >> 4, col4 = (c & 15) << 2;
      const unsigned ldsoff = (unsigned)((r * DH + col4) * 4);
      const unsigned goff   = (unsigned)((((j0 + r) * D_MODEL) + col4) * 4);
      async_copy_b128(k_lds0 + ldsoff, goff, kb);
      async_copy_b128(v_lds0 + ldsoff, goff, vb);
    }
    wait_async0();
    __syncthreads();
    v16h a;
    #pragma unroll
    for (int s = 0; s < 16; ++s) {
      const int kk = (s & 7) + ((s >> 3) << 4) + (hi << 3);
      const int j  = j0 + kk;
      float l1 = 0.0f;
      #pragma unroll 8
      for (int d = 0; d < DH; ++d) l1 += fabsf(q_lds[m][d] - k_lds[kk][d]);
      const float dsq  = l1 * l1 * (1.0f / 64.0f);    // (l1/sqrt(Dh))^2
      const float base = 1.0f / (1.0f + gp * dsq + 1e-6f);
      float aw = __expf(rp * __logf(base));
      if (j > qi) aw = 0.0f;                          // causal mask
      den += aw;
      a[s] = (_Float16)aw;
    }
    v16h b0, b1, b2, b3;
    #pragma unroll
    for (int s = 0; s < 16; ++s) {
      const int kk2 = (s & 7) + ((s >> 3) << 4) + (hi << 3);
      b0[s] = (_Float16)v_lds[kk2][ 0 + m];
      b1[s] = (_Float16)v_lds[kk2][16 + m];
      b2[s] = (_Float16)v_lds[kk2][32 + m];
      b3[s] = (_Float16)v_lds[kk2][48 + m];
    }
    c0 = __builtin_amdgcn_wmma_f32_16x16x32_f16(false, a, false, b0, (short)0, c0, false, false);
    c1 = __builtin_amdgcn_wmma_f32_16x16x32_f16(false, a, false, b1, (short)0, c1, false, false);
    c2 = __builtin_amdgcn_wmma_f32_16x16x32_f16(false, a, false, b2, (short)0, c2, false, false);
    c3 = __builtin_amdgcn_wmma_f32_16x16x32_f16(false, a, false, b3, (short)0, c3, false, false);
  }
  __syncthreads();
  denom[lane] = den;
  __syncthreads();
  if (lane < 16) denom[lane] += denom[lane + 16];
  __syncthreads();
  float* ob = out + (size_t)(b * SEQ) * D_MODEL + h * DH;
  #pragma unroll
  for (int r = 0; r < 8; ++r) {
    const int M = (hi ? 8 : 0) + r;
    const float inv = 1.0f / (denom[M] + 1e-6f);
    const size_t ro = (size_t)(i0 + M) * D_MODEL;
    ob[ro +  0 + m] = c0[r] * inv;
    ob[ro + 16 + m] = c1[r] * inv;
    ob[ro + 32 + m] = c2[r] * inv;
    ob[ro + 48 + m] = c3[r] * inv;
  }
}

// ---------------- x_res = x + attn @ wo^T + wo_b (WMMA f16) -----------------
__global__ void wo_gemm_kernel(const float* __restrict__ attn, const float* __restrict__ wo,
                               const float* __restrict__ wob, const float* __restrict__ xin,
                               float* __restrict__ xres) {
  __shared__ _Float16 a_lds[16][34];
  __shared__ _Float16 b_lds[16][34];
  const int lane = threadIdx.x;
  const int m0 = blockIdx.x * 16, n0 = blockIdx.y * 16;
  const int m = lane & 15, hi = lane >> 4;
  v8f c = {};
  for (int k0 = 0; k0 < D_MODEL; k0 += 32) {
    __syncthreads();
    for (int i = lane; i < 16 * 32; i += 32) {
      int r = i >> 5, cc = i & 31;
      a_lds[r][cc] = (_Float16)attn[(size_t)(n0 + r) * D_MODEL + k0 + cc];
      b_lds[r][cc] = (_Float16)wo[(size_t)(m0 + r) * D_MODEL + k0 + cc];
    }
    __syncthreads();
    v16h a, bb;
    #pragma unroll
    for (int s = 0; s < 16; ++s) {
      const int kk  = (s & 7) + ((s >> 3) << 4) + (hi << 3);
      const int kk2 = (s & 7) + ((s >> 3) << 4) + (hi << 3);
      a[s]  = a_lds[m][kk];
      bb[s] = b_lds[m][kk2];   // B_wmma[K][N] = wo[m0+N][K]
    }
    c = __builtin_amdgcn_wmma_f32_16x16x32_f16(false, a, false, bb, (short)0, c, false, false);
  }
  #pragma unroll
  for (int r = 0; r < 8; ++r) {
    const int n  = n0 + ((hi ? 8 : 0) + r);
    const int mm = m0 + m;
    xres[(size_t)n * D_MODEL + mm] = xin[(size_t)n * D_MODEL + mm] + c[r] + wob[mm];
  }
}

// ---------------- tgv = tanh(gate_silu * val)  (writes over val) ------------
__global__ void gv_kernel(const float* __restrict__ g, float* __restrict__ v, int n) {
  for (int i = blockIdx.x * blockDim.x + threadIdx.x; i < n; i += gridDim.x * blockDim.x)
    v[i] = tanhf(g[i] * v[i]);
}

// ---------------- y = x_res + f + 0.5*|shift(h2) - sigmoid(dir_w)| ----------
__global__ void final_kernel(const float* __restrict__ xres, const float* __restrict__ f,
                             const float* __restrict__ h2, const float* __restrict__ dirw,
                             float* __restrict__ y) {
  for (int i = blockIdx.x * blockDim.x + threadIdx.x; i < NROWS * D_MODEL;
       i += gridDim.x * blockDim.x) {
    const int n = i / D_MODEL, d = i % D_MODEL;
    const int t = n % SEQ;
    const float xp = (t == 0) ? 0.0f : h2[(size_t)(n - 1) * D_MODEL + d];
    const float dv = fabsf(xp - sigmoid_f(dirw[d]));
    y[i] = xres[i] + f[i] + 0.5f * dv;
  }
}

extern "C" void kernel_launch(void* const* d_in, const int* in_sizes, int n_in,
                              void* d_out, int out_size, void* d_ws, size_t ws_size,
                              hipStream_t stream) {
  (void)in_sizes; (void)n_in; (void)out_size; (void)ws_size;
  const float* x      = (const float*)d_in[0];
  const float* ln1_w  = (const float*)d_in[1];
  const float* ln1_b  = (const float*)d_in[2];
  const float* wq_w   = (const float*)d_in[3];
  const float* wq_g   = (const float*)d_in[4];
  const float* wk_w   = (const float*)d_in[5];
  const float* wk_g   = (const float*)d_in[6];
  const float* wv_w   = (const float*)d_in[7];
  const float* wv_g   = (const float*)d_in[8];
  const float* wo_w   = (const float*)d_in[9];
  const float* wo_b   = (const float*)d_in[10];
  const float* gamma  = (const float*)d_in[11];
  const float* rho    = (const float*)d_in[12];
  const float* ln2_w  = (const float*)d_in[13];
  const float* ln2_b  = (const float*)d_in[14];
  const float* gate_w = (const float*)d_in[15];
  const float* gate_g = (const float*)d_in[16];
  const float* val_w  = (const float*)d_in[17];
  const float* val_g  = (const float*)d_in[18];
  const float* out_w  = (const float*)d_in[19];
  const float* out_g  = (const float*)d_in[20];
  const float* dir_w  = (const float*)d_in[21];
  float* y = (float*)d_out;

  const size_t ND = (size_t)NROWS * D_MODEL;   // 1M
  const size_t DD = (size_t)D_MODEL * D_MODEL; // 256K
  const size_t HD = (size_t)HIDDEN * D_MODEL;  // 1M
  const size_t NH = (size_t)NROWS * HIDDEN;    // 4M
  float* ws = (float*)d_ws;
  float* A0 = ws;            // tx (tanh of ln out), reused for both halves
  float* A1 = A0 + ND;       // q
  float* A2 = A1 + ND;       // k
  float* A3 = A2 + ND;       // v
  float* A4 = A3 + ND;       // attn_out, later f
  float* A5 = A4 + ND;       // x_res
  float* A6 = A5 + ND;       // ln scratch / h2
  float* WT = A6 + ND;       // tanh'd weights region (2*HD floats)
  float* G  = WT + 2 * HD;   // gate  [N, HIDDEN]
  float* V  = G + NH;        // val -> tgv [N, HIDDEN]

  // ---- ln1 -> tanh -> tx
  ln512_kernel<<<NROWS, 256, 0, stream>>>(x, ln1_w, ln1_b, A6);
  tanh_kernel<<<2048, 256, 0, stream>>>(A6, A0, (int)ND);
  // ---- tanh QKV weights
  tanh_kernel<<<1024, 256, 0, stream>>>(wq_w, WT,          (int)DD);
  tanh_kernel<<<1024, 256, 0, stream>>>(wk_w, WT + DD,     (int)DD);
  tanh_kernel<<<1024, 256, 0, stream>>>(wv_w, WT + 2 * DD, (int)DD);
  // ---- proxy q/k/v
  dim3 pb(16, 16);
  dim3 pgD(D_MODEL / 16, NROWS / 16);
  proxy_kernel<<<pgD, pb, 0, stream>>>(A0, WT,          wq_g, A1, NROWS, D_MODEL, D_MODEL, 1);
  proxy_kernel<<<pgD, pb, 0, stream>>>(A0, WT + DD,     wk_g, A2, NROWS, D_MODEL, D_MODEL, 1);
  proxy_kernel<<<pgD, pb, 0, stream>>>(A0, WT + 2 * DD, wv_g, A3, NROWS, D_MODEL, D_MODEL, 0);
  // ---- attention (streaming, async-LDS staging + WMMA PV)
  attn_kernel<<<dim3(SEQ / 16, BATCH * HEADS), 32, 0, stream>>>(A1, A2, A3, gamma, rho, A4);
  // ---- x_res = x + attn @ wo^T + wo_b (WMMA)
  wo_gemm_kernel<<<dim3(D_MODEL / 16, NROWS / 16), 32, 0, stream>>>(A4, wo_w, wo_b, x, A5);
  // ---- ln2 -> h2, tanh -> tx2
  ln512_kernel<<<NROWS, 256, 0, stream>>>(A5, ln2_w, ln2_b, A6);
  tanh_kernel<<<2048, 256, 0, stream>>>(A6, A0, (int)ND);
  // ---- MLP proxies
  tanh_kernel<<<4096, 256, 0, stream>>>(gate_w, WT,      (int)HD);
  tanh_kernel<<<4096, 256, 0, stream>>>(val_w,  WT + HD, (int)HD);
  dim3 pgH(HIDDEN / 16, NROWS / 16);
  proxy_kernel<<<pgH, pb, 0, stream>>>(A0, WT,      gate_g, G, NROWS, HIDDEN, D_MODEL, 2);
  proxy_kernel<<<pgH, pb, 0, stream>>>(A0, WT + HD, val_g,  V, NROWS, HIDDEN, D_MODEL, 0);
  gv_kernel<<<4096, 256, 0, stream>>>(G, V, (int)NH);
  tanh_kernel<<<4096, 256, 0, stream>>>(out_w, WT, (int)HD);
  proxy_kernel<<<pgD, pb, 0, stream>>>(V, WT, out_g, A4, NROWS, D_MODEL, HIDDEN, 0);
  // ---- final combine
  final_kernel<<<2048, 256, 0, stream>>>(A5, A4, A6, dir_w, y);
}
